// softmaxattention1_78512002171064
// MI455X (gfx1250) — matrix operand order repaired and checked by reading, not compiled, not run on hardware
//
#include <hip/hip_runtime.h>
#include <hip/hip_bf16.h>

// ---------------------------------------------------------------------------
// Types for CDNA5 WMMA
// ---------------------------------------------------------------------------
typedef __attribute__((ext_vector_type(16))) __bf16 v16bf;
typedef __attribute__((ext_vector_type(8)))  float  v8f;

#define TM 64            // block tile M
#define TN 128           // block tile N
#define KT 32            // k step
#define LDS_STRIDE 40    // rows x 40 bf16 (pad 8)

union FragU { uint4 u[2]; v16bf v; };

__device__ __forceinline__ unsigned short f2bf(float f) {
  unsigned u = __float_as_uint(f);
  unsigned r = u + 0x7FFFu + ((u >> 16) & 1u);   // round-to-nearest-even
  return (unsigned short)(r >> 16);
}

// A-matrix fragment (16x32 bf16, ISA 7.12.2): lanes 0-15 hold row M=lane,
// K = {0..7, 16..23}; lanes 16-31 hold K = {8..15, 24..31}.
__device__ __forceinline__ v16bf ldsFragA(const unsigned short* t, int row) {
  int lane = threadIdx.x & 31;
  int r  = row + (lane & 15);
  int kb = (lane >> 4) << 3;                  // 0 or 8
  FragU f;
  f.u[0] = *(const uint4*)(t + r * LDS_STRIDE + kb);
  f.u[1] = *(const uint4*)(t + r * LDS_STRIDE + kb + 16);
  return f.v;
}

// B-matrix fragment (32x16 bf16): lane n (0-15) holds column N=n, K={0..7,16..23};
// lanes 16-31 hold K={8..15,24..31} (same as A). Tile is stored N-major in LDS.
__device__ __forceinline__ v16bf ldsFragB(const unsigned short* t, int row) {
  int lane = threadIdx.x & 31;
  int r  = row + (lane & 15);
  int kb = (lane >> 4) << 3;                  // 0 or 8
  FragU f;
  f.u[0] = *(const uint4*)(t + r * LDS_STRIDE + kb);
  f.u[1] = *(const uint4*)(t + r * LDS_STRIDE + kb + 16);
  return f.v;
}

// ---------------------------------------------------------------------------
// Workhorse GEMM: out[M,N] = A[M,K]*B[K,N] (+bias[m]) (+addSrc), bf16 in,
// fp32 accumulate. Block tile 64x128, 4 waves of 32x64 (2x4 WMMA accums).
// Staging issues all global loads before LDS stores; next k-tile is touched
// with an unconditional speculative global_prefetch_b8 (safe past the end).
// If ATRANS, A element (m,k) is read from A[k*lda + m]  (c = y2^T * y1).
// ---------------------------------------------------------------------------
template <bool ATRANS, bool HAS_BIAS, bool HAS_ADD, bool OUT_F, bool OUT_B>
__global__ __launch_bounds__(128)
void gemm_wmma_bf16(const unsigned short* __restrict__ A,
                    const unsigned short* __restrict__ Bm,
                    const float* __restrict__ bias,
                    const float* __restrict__ addSrc,
                    float* __restrict__ outF,
                    unsigned short* __restrict__ outB,
                    int K, int lda, int ldb, int ldc,
                    long aBatch, long bBatch, long cBatch) {
  __shared__ unsigned short As[TM * LDS_STRIDE];
  __shared__ unsigned short Bs[TN * LDS_STRIDE];

  const int z = blockIdx.z;
  A  += (long)z * aBatch;
  Bm += (long)z * bBatch;
  const long cOff = (long)z * cBatch;

  const int nBase = blockIdx.x * TN;
  const int mBase = blockIdx.y * TM;

  const int tid  = threadIdx.x;
  const int wave = tid >> 5;
  const int wm   = (wave >> 1) * 32;     // 0 or 32
  const int wn   = (wave & 1) * 64;      // 0 or 64

  // Per-thread staging coordinates (fixed across k-steps).
  // A tile (64 x 32): 2 vectors of 8 bf16 per thread.
  int amRow[2], amK[2];
  #pragma unroll
  for (int it = 0; it < 2; ++it) {
    int idx = tid + it * 128;            // 0..255
    if constexpr (!ATRANS) { amRow[it] = idx >> 2; amK[it] = (idx & 3) << 3; }
    else                   { amK[it]  = idx >> 3; amRow[it] = (idx & 7) << 3; }
  }
  // B tile (32 x 128): 4 vectors of 8 bf16 per thread.
  int bk[4], bn[4];
  #pragma unroll
  for (int it = 0; it < 4; ++it) {
    int idx = tid + it * 128;            // 0..511
    bk[it] = idx >> 4;                   // 0..31
    bn[it] = (idx & 15) << 3;            // 0..120
  }

  v8f acc[2][4] = {};

  for (int k0 = 0; k0 < K; k0 += KT) {
    // ---- phase 1: issue all global loads (LOADcnt batches up) ----
    uint4 va[2], vb[4];
    #pragma unroll
    for (int it = 0; it < 2; ++it) {
      const unsigned short* p = ATRANS
          ? A + (long)(k0 + amK[it]) * lda + mBase + amRow[it]
          : A + (long)(mBase + amRow[it]) * lda + k0 + amK[it];
      va[it] = *(const uint4*)p;
    }
    #pragma unroll
    for (int it = 0; it < 4; ++it)
      vb[it] = *(const uint4*)(Bm + (long)(k0 + bk[it]) * ldb + nBase + bn[it]);

    // ---- speculative prefetch of next k-tile (safe past the end) ----
    #pragma unroll
    for (int it = 0; it < 2; ++it) {
      const unsigned short* p = ATRANS
          ? A + (long)(k0 + KT + amK[it]) * lda + mBase + amRow[it]
          : A + (long)(mBase + amRow[it]) * lda + k0 + KT + amK[it];
      __builtin_prefetch(p, 0, 3);
    }
    #pragma unroll
    for (int it = 0; it < 4; ++it)
      __builtin_prefetch(Bm + (long)(k0 + KT + bk[it]) * ldb + nBase + bn[it], 0, 3);

    // ---- phase 2: LDS stores ----
    if constexpr (!ATRANS) {
      #pragma unroll
      for (int it = 0; it < 2; ++it)
        *(uint4*)(&As[amRow[it] * LDS_STRIDE + amK[it]]) = va[it];
    } else {
      #pragma unroll
      for (int it = 0; it < 2; ++it) {
        const unsigned short* s = (const unsigned short*)&va[it];
        #pragma unroll
        for (int j = 0; j < 8; ++j) As[(amRow[it] + j) * LDS_STRIDE + amK[it]] = s[j];
      }
    }
    #pragma unroll
    for (int it = 0; it < 4; ++it) {
      const unsigned short* s = (const unsigned short*)&vb[it];
      #pragma unroll
      for (int j = 0; j < 8; ++j) Bs[(bn[it] + j) * LDS_STRIDE + bk[it]] = s[j];
    }
    __syncthreads();

    // ---- phase 3: WMMA ----
    v16bf a0 = ldsFragA(As, wm);
    v16bf a1 = ldsFragA(As, wm + 16);
    #pragma unroll
    for (int j = 0; j < 4; ++j) {
      v16bf b = ldsFragB(Bs, wn + j * 16);
      acc[0][j] = __builtin_amdgcn_wmma_f32_16x16x32_bf16(false, a0, false, b, (short)0, acc[0][j], false, false);
      acc[1][j] = __builtin_amdgcn_wmma_f32_16x16x32_bf16(false, a1, false, b, (short)0, acc[1][j], false, false);
    }
    __syncthreads();
  }

  // ---- epilogue: C/D layout = VGPR r: lanes0-15 (M=r,N=lane), lanes16-31
  // (M=8+r, N=lane-16). Branch-free (all options compile-time). ----
  const int lane   = tid & 31;
  const int col    = lane & 15;
  const int rowOff = (lane >> 4) << 3;
  #pragma unroll
  for (int i = 0; i < 2; ++i) {
    #pragma unroll
    for (int j = 0; j < 4; ++j) {
      int m0 = mBase + wm + i * 16 + rowOff;
      int n  = nBase + wn + j * 16 + col;
      long base = cOff + (long)m0 * ldc + n;
      #pragma unroll
      for (int r = 0; r < 8; ++r) {
        float v = acc[i][j][r];
        if constexpr (HAS_BIAS) v += bias[m0 + r];
        long o = base + (long)r * ldc;
        if constexpr (HAS_ADD) v += addSrc[o];
        if constexpr (OUT_F) outF[o] = v;
        if constexpr (OUT_B) outB[o] = f2bf(v);
      }
    }
  }
}

// ---------------------------------------------------------------------------
// Elementwise / reduction kernels
// ---------------------------------------------------------------------------
__global__ void k_f32_to_bf16(const float* __restrict__ src,
                              unsigned short* __restrict__ dst, long n) {
  long i = (long)blockIdx.x * blockDim.x + threadIdx.x;
  long st = (long)gridDim.x * blockDim.x;
  for (; i < n; i += st) dst[i] = f2bf(src[i]);
}

// ye[b,hw] = dot(W6, x[b,:,hw]) + b6
__global__ __launch_bounds__(256) void k_ye(const float* __restrict__ x,
                                            const float* __restrict__ W6,
                                            const float* __restrict__ b6,
                                            float* __restrict__ ye) {
  int pix = blockIdx.x * 256 + threadIdx.x;      // < 16384
  int b = pix >> 10, hw = pix & 1023;
  const float* xp = x + ((long)b * 512) * 1024 + hw;
  float s = b6[0];
  for (int c = 0; c < 512; ++c) s += W6[c] * xp[(long)c * 1024];
  ye[pix] = s;
}

// softmax over width (groups of 32) in-place
__global__ void k_softmax_w(float* __restrict__ ye) {
  int row = blockIdx.x * blockDim.x + threadIdx.x;   // < 512 rows (b*h)
  if (row >= 512) return;
  float* p = ye + row * 32;
  float mx = -3.4e38f;
  float e[32];
  for (int i = 0; i < 32; ++i) mx = fmaxf(mx, p[i]);
  float s = 0.f;
  for (int i = 0; i < 32; ++i) { e[i] = expf(p[i] - mx); s += e[i]; }
  float inv = 1.f / s;
  for (int i = 0; i < 32; ++i) p[i] = e[i] * inv;
}

// gb = bf16( ye_[b,hw] * y3f[b,c,hw] )
__global__ void k_gate(const float* __restrict__ ye, const float* __restrict__ y3f,
                       unsigned short* __restrict__ gb, long n) {
  long i = (long)blockIdx.x * blockDim.x + threadIdx.x;
  long st = (long)gridDim.x * blockDim.x;
  for (; i < n; i += st) {
    long hw = i & 1023;
    long b  = i >> 19;                  // C*HW = 2^19
    gb[i] = f2bf(ye[(b << 10) + hw] * y3f[i]);
  }
}

__global__ void k_zero(float* p, int n) {
  int i = blockIdx.x * blockDim.x + threadIdx.x;
  if (i < n) p[i] = 0.f;
}

// per-batch sum / sumsq over C*HW (GroupNorm pass 1)
__global__ __launch_bounds__(256) void k_gn_reduce(const float* __restrict__ t,
                                                   float* __restrict__ sums) {
  __shared__ float s1[256], s2[256];
  int b = blockIdx.y;
  const float* p = t + (long)b * 524288 + (long)blockIdx.x * 8192;
  float a = 0.f, q = 0.f;
  for (int i = threadIdx.x; i < 8192; i += 256) { float v = p[i]; a += v; q += v * v; }
  s1[threadIdx.x] = a; s2[threadIdx.x] = q; __syncthreads();
  for (int s = 128; s > 0; s >>= 1) {
    if (threadIdx.x < s) { s1[threadIdx.x] += s1[threadIdx.x + s]; s2[threadIdx.x] += s2[threadIdx.x + s]; }
    __syncthreads();
  }
  if (threadIdx.x == 0) { atomicAdd(&sums[b * 2], s1[0]); atomicAdd(&sums[b * 2 + 1], s2[0]); }
}

// GroupNorm pass 2 -> bf16
__global__ void k_gn_apply(const float* __restrict__ t, const float* __restrict__ sums,
                           const float* __restrict__ gamma, const float* __restrict__ beta,
                           unsigned short* __restrict__ tnb, long n) {
  long i = (long)blockIdx.x * blockDim.x + threadIdx.x;
  long st = (long)gridDim.x * blockDim.x;
  for (; i < n; i += st) {
    long b = i >> 19;
    int  c = (int)((i >> 10) & 511);
    float mu  = sums[b * 2] * (1.f / 524288.f);
    float var = sums[b * 2 + 1] * (1.f / 524288.f) - mu * mu;
    float v = (t[i] - mu) * rsqrtf(var + 1e-5f) * gamma[c] + beta[c];
    tnb[i] = f2bf(v);
  }
}

// row softmax over 1024-wide rows of cmat -> bf16
__global__ __launch_bounds__(256) void k_rowsoftmax(const float* __restrict__ cm,
                                                    unsigned short* __restrict__ cw) {
  __shared__ float red[256];
  long base = (long)blockIdx.x * 1024;
  float v[4];
  #pragma unroll
  for (int i = 0; i < 4; ++i) v[i] = cm[base + threadIdx.x + i * 256];
  float mx = fmaxf(fmaxf(v[0], v[1]), fmaxf(v[2], v[3]));
  red[threadIdx.x] = mx; __syncthreads();
  for (int s = 128; s > 0; s >>= 1) {
    if (threadIdx.x < s) red[threadIdx.x] = fmaxf(red[threadIdx.x], red[threadIdx.x + s]);
    __syncthreads();
  }
  mx = red[0]; __syncthreads();
  float s = 0.f;
  #pragma unroll
  for (int i = 0; i < 4; ++i) { v[i] = expf(v[i] - mx); s += v[i]; }
  red[threadIdx.x] = s; __syncthreads();
  for (int s2 = 128; s2 > 0; s2 >>= 1) {
    if (threadIdx.x < s2) red[threadIdx.x] += red[threadIdx.x + s2];
    __syncthreads();
  }
  float inv = 1.f / red[0];
  #pragma unroll
  for (int i = 0; i < 4; ++i) cw[base + threadIdx.x + i * 256] = f2bf(v[i] * inv);
}

// avg[b,c] = mean over HW of y_last_a
__global__ __launch_bounds__(256) void k_avgpool(const float* __restrict__ yl,
                                                 float* __restrict__ avg) {
  __shared__ float red[256];
  long base = (long)blockIdx.x * 1024;
  float s = 0.f;
  for (int i = threadIdx.x; i < 1024; i += 256) s += yl[base + i];
  red[threadIdx.x] = s; __syncthreads();
  for (int k = 128; k > 0; k >>= 1) {
    if (threadIdx.x < k) red[threadIdx.x] += red[threadIdx.x + k];
    __syncthreads();
  }
  if (threadIdx.x == 0) avg[blockIdx.x] = red[0] * (1.f / 1024.f);
}

// fused head: y4 = W4*avg+b4; y4 /= max(||y4||,1e-12); out = W5*y4+b5
__global__ __launch_bounds__(512) void k_head(const float* __restrict__ avg,
                                              const float* __restrict__ W4, const float* __restrict__ b4,
                                              const float* __restrict__ W5, const float* __restrict__ b5,
                                              float* __restrict__ out) {
  __shared__ float sIn[512], sMid[512], sRed[512];
  int b = blockIdx.x, o = threadIdx.x;
  sIn[o] = avg[b * 512 + o];
  __syncthreads();
  float y4 = b4[o];
  const float* w = W4 + (long)o * 512;
  for (int c = 0; c < 512; ++c) y4 += w[c] * sIn[c];
  sRed[o] = y4 * y4; __syncthreads();
  for (int s = 256; s > 0; s >>= 1) {
    if (o < s) sRed[o] += sRed[o + s];
    __syncthreads();
  }
  float nrm = sqrtf(sRed[0]);
  sMid[o] = y4 / fmaxf(nrm, 1e-12f);
  __syncthreads();
  float y5 = b5[o];
  const float* w5 = W5 + (long)o * 512;
  for (int c = 0; c < 512; ++c) y5 += w5[c] * sMid[c];
  out[b * 512 + o] = y5;
}

// ---------------------------------------------------------------------------
// Launcher
// ---------------------------------------------------------------------------
extern "C" void kernel_launch(void* const* d_in, const int* in_sizes, int n_in,
                              void* d_out, int out_size, void* d_ws, size_t ws_size,
                              hipStream_t stream) {
  (void)in_sizes; (void)n_in; (void)out_size; (void)ws_size;
  const float* x     = (const float*)d_in[0];
  const float* W1    = (const float*)d_in[1];  const float* b1 = (const float*)d_in[2];
  const float* W2    = (const float*)d_in[3];  const float* b2 = (const float*)d_in[4];
  const float* W3    = (const float*)d_in[5];  const float* b3 = (const float*)d_in[6];
  const float* W4    = (const float*)d_in[7];  const float* b4 = (const float*)d_in[8];
  const float* W5    = (const float*)d_in[9];  const float* b5 = (const float*)d_in[10];
  const float* W6    = (const float*)d_in[11]; const float* b6 = (const float*)d_in[12];
  const float* W7    = (const float*)d_in[13]; const float* b7 = (const float*)d_in[14];
  const float* W8    = (const float*)d_in[15]; const float* b8 = (const float*)d_in[16];
  const float* gamma = (const float*)d_in[17]; const float* beta = (const float*)d_in[18];

  const long PB = 512L * 1024;       // per-batch C*HW
  const long NX = 16L * PB;          // 8,388,608
  const long NC = 16L * 1024 * 1024; // B*HW*HW

  char* ws = (char*)d_ws;
  size_t off = 0;
  auto take = [&](size_t bytes) -> void* {
    void* p = ws + off;
    off = (off + bytes + 255) & ~(size_t)255;
    return p;
  };
  unsigned short* xb   = (unsigned short*)take(NX * 2);
  unsigned short* w1b  = (unsigned short*)take(512 * 512 * 2);
  unsigned short* w2b  = (unsigned short*)take(512 * 512 * 2);
  unsigned short* w3b  = (unsigned short*)take(512 * 512 * 2);
  unsigned short* w7b  = (unsigned short*)take(512 * 512 * 2);
  unsigned short* w8b  = (unsigned short*)take(512 * 512 * 2);
  unsigned short* y1b  = (unsigned short*)take(NX * 2);
  unsigned short* y2b  = (unsigned short*)take(NX * 2);
  float*          y3f  = (float*)take(NX * 4);
  unsigned short* y3b  = (unsigned short*)take(NX * 2);
  float*          yev  = (float*)take(16384 * 4);
  unsigned short* gb   = (unsigned short*)take(NX * 2);
  float*          tf   = (float*)take(NX * 4);
  unsigned short* tnb  = gb;                     // reuse: gb dead after t GEMM
  float*          yeef = (float*)take(NX * 4);
  float*          cmat = (float*)take(NC * 4);
  unsigned short* cwb  = (unsigned short*)take(NC * 2);
  float*          ylast = cmat;                  // reuse: cmat dead after softmax
  float*          avg  = (float*)take(8192 * 4);
  float*          sums = (float*)take(32 * 4);

  // --- bf16 operand conversion ---
  k_f32_to_bf16<<<1024, 256, 0, stream>>>(x,  xb,  NX);
  k_f32_to_bf16<<<256, 256, 0, stream>>>(W1, w1b, 262144);
  k_f32_to_bf16<<<256, 256, 0, stream>>>(W2, w2b, 262144);
  k_f32_to_bf16<<<256, 256, 0, stream>>>(W3, w3b, 262144);
  k_f32_to_bf16<<<256, 256, 0, stream>>>(W7, w7b, 262144);
  k_f32_to_bf16<<<256, 256, 0, stream>>>(W8, w8b, 262144);

  dim3 gConv(1024 / TN, 512 / TM, 16);    // (8, 8, 16)
  dim3 gAtt (1024 / TN, 1024 / TM, 16);   // (8, 16, 16)

  // y1 = W1*x+b1 (bf16 only), y2 likewise, y3 (f32 + bf16)
  gemm_wmma_bf16<false, true, false, false, true><<<gConv, 128, 0, stream>>>(
      w1b, xb, b1, nullptr, nullptr, y1b, 512, 512, 1024, 1024, 0, PB, PB);
  gemm_wmma_bf16<false, true, false, false, true><<<gConv, 128, 0, stream>>>(
      w2b, xb, b2, nullptr, nullptr, y2b, 512, 512, 1024, 1024, 0, PB, PB);
  gemm_wmma_bf16<false, true, false, true, true><<<gConv, 128, 0, stream>>>(
      w3b, xb, b3, nullptr, y3f, y3b, 512, 512, 1024, 1024, 0, PB, PB);

  // ye = W6*x+b6, softmax over width, gate = ye_*y3
  k_ye<<<64, 256, 0, stream>>>(x, W6, b6, yev);
  k_softmax_w<<<2, 256, 0, stream>>>(yev);
  k_gate<<<2048, 256, 0, stream>>>(yev, y3f, gb, NX);

  // t = W7*gate+b7 ; GroupNorm(1 group) -> tnb ; yee = W8*tn+b8
  gemm_wmma_bf16<false, true, false, true, false><<<gConv, 128, 0, stream>>>(
      w7b, gb, b7, nullptr, tf, nullptr, 512, 512, 1024, 1024, 0, PB, PB);
  k_zero<<<1, 32, 0, stream>>>(sums, 32);
  k_gn_reduce<<<dim3(64, 16), 256, 0, stream>>>(tf, sums);
  k_gn_apply<<<2048, 256, 0, stream>>>(tf, sums, gamma, beta, tnb, NX);
  gemm_wmma_bf16<false, true, false, true, false><<<gConv, 128, 0, stream>>>(
      w8b, tnb, b8, nullptr, yeef, nullptr, 512, 512, 1024, 1024, 0, PB, PB);

  // c[b,n,m] = sum_c y2[c,n]*y1[c,m]  (A = y2 read transposed)
  gemm_wmma_bf16<true, false, false, true, false><<<gAtt, 128, 0, stream>>>(
      y2b, y1b, nullptr, nullptr, cmat, nullptr, 512, 1024, 1024, 1024, PB, PB, 1024L * 1024);
  k_rowsoftmax<<<16384, 256, 0, stream>>>(cmat, cwb);

  // y_last = y3 * Cw + yee_
  gemm_wmma_bf16<false, false, true, true, false><<<gConv, 128, 0, stream>>>(
      y3b, cwb, nullptr, yeef, ylast, nullptr, 1024, 1024, 1024, 1024, PB, 1024L * 1024, PB);

  // avg pool -> head (GEMV, L2-normalize, GEMV) -> d_out [16,512]
  k_avgpool<<<8192, 256, 0, stream>>>(ylast, avg);
  k_head<<<16, 512, 0, stream>>>(avg, W4, b4, W5, b5, (float*)d_out);
}
